// DeformableTemporalAttention_51221779972754
// MI455X (gfx1250) — matrix operand order repaired and checked by reading, not compiled, not run on hardware
//
#include <hip/hip_runtime.h>

// ---------------------------------------------------------------------------
// DeformableTemporalAttention for MI455X (gfx1250, wave32, WMMA)
// - all dense contractions via v_wmma_f32_16x16x32_bf16 (f32 accumulate)
// - A-tiles staged global->LDS with GLOBAL_LOAD_ASYNC_TO_LDS_B128 (ASYNCcnt)
// - A fragments hoisted to registers so ds_loads clause + WMMAs run back-to-back
// ---------------------------------------------------------------------------

typedef __attribute__((ext_vector_type(16))) __bf16 v16bf;
typedef __attribute__((ext_vector_type(8)))  __bf16 v8bf;
typedef __attribute__((ext_vector_type(8)))  float  v8f;

#define Bq   2
#define Nv   4
#define Cc   128
#define HEADS 8
#define HD   16
#define Pp   4
#define Hh   96
#define Ww   96
#define HW   (Hh*Ww)          // 9216
#define PIX  (Bq*HW)          // 18432
#define PIXV (Bq*Nv*HW)       // 73728
#define ROWS (Bq*HEADS*HW)    // 147456

__device__ __forceinline__ float gelu_exact(float x) {
    return 0.5f * x * (1.0f + erff(x * 0.70710678118654752f));
}
__device__ __forceinline__ float sigm(float x) {
    return 1.0f / (1.0f + __expf(-x));
}
__device__ __forceinline__ int iclampi(int v, int lo, int hi) {
    return v < lo ? lo : (v > hi ? hi : v);
}
__device__ __forceinline__ void decode_m(int m, int& b, int& hh, int& y, int& x) {
    x = m % Ww; int t = m / Ww; y = t % Hh; t /= Hh; hh = t & (HEADS - 1); b = t >> 3;
}

// async 16-byte global->LDS copy (per-lane), tracked by ASYNCcnt.
// ldsoff: low 32 bits of a flat LDS pointer == LDS byte offset (ISA 10.2).
__device__ __forceinline__ void async_g2l_b128(unsigned ldsoff, const void* gaddr) {
    asm volatile("global_load_async_to_lds_b128 %0, %1, off"
                 :: "v"(ldsoff), "v"(gaddr) : "memory");
}
__device__ __forceinline__ void wait_asynccnt0() {
    asm volatile("s_wait_asynccnt 0x0" ::: "memory");
}

// ------------------------- weight / activation packing ---------------------

// pack 128x128 f32 weight w[o][c] into WMMA-B fragment order:
// [jtile(8)][ktile(4)][lane(32)][half(16)]
__global__ void pack_w128(const float* __restrict__ w, __bf16* __restrict__ dst) {
    int tid = blockIdx.x * blockDim.x + threadIdx.x;
    if (tid >= 8 * 4 * 32 * 16) return;
    int h = tid & 15, lane = (tid >> 4) & 31, t = (tid >> 9) & 3, j = tid >> 11;
    int n = j * 16 + (lane & 15);
    int k = t * 32 + ((h >> 3) << 4) + ((lane >> 4) << 3) + (h & 7);
    dst[tid] = (__bf16)w[n * 128 + k];
}

// combined GRU weight 96x64: rows 0..47 = wih(48x18) in cols 0..17,
// rows 48..95 = whh(48x16) in cols 18..33; fragment order [j(6)][t(2)][lane][16]
__global__ void pack_wg(const float* __restrict__ wih, const float* __restrict__ whh,
                        __bf16* __restrict__ dst) {
    int tid = blockIdx.x * blockDim.x + threadIdx.x;
    if (tid >= 6 * 2 * 32 * 16) return;
    int h = tid & 15, lane = (tid >> 4) & 31, t = (tid >> 9) & 1, j = tid >> 10;
    int g = j * 16 + (lane & 15);
    int k = t * 32 + ((h >> 3) << 4) + ((lane >> 4) << 3) + (h & 7);
    float v = 0.0f;
    if (g < 48) { if (k < 18) v = wih[g * 18 + k]; }
    else        { if (k >= 18 && k < 34) v = whh[(g - 48) * 16 + (k - 18)]; }
    dst[tid] = (__bf16)v;
}

// NCHW f32 -> pixel-major [batch*HW, 128] bf16
__global__ void pack_act(const float* __restrict__ src, __bf16* __restrict__ dst, int batch) {
    int tid = blockIdx.x * blockDim.x + threadIdx.x;
    int total = batch * Cc * HW;
    if (tid >= total) return;
    int p = tid % HW; int t = tid / HW; int c = t % Cc; int mb = t / Cc;
    dst[((size_t)mb * HW + p) * Cc + c] = (__bf16)src[tid];
}

// --------------------------- WMMA 128->128 GEMM ----------------------------
// Y[M,128] = epilogue( X[M,128](bf16) @ W^T + bias ), 64 rows per block.
// block = (32,8): wave j owns output columns j*16..j*16+15; B fragments are
// loaded once and reused over 4 M-subtiles; the 64x128 A-tile (16 KB) is
// staged into LDS with async global->LDS loads shared by all 8 waves.
// A fragments are hoisted to registers before the WMMA burst so the 8
// ds_load_b128 can clause behind one DScnt wait.
// mode 0: f32 row-major; mode 1: GELU->bf16 row-major; mode 2: *sigmoid(gate)
// with f32 NCHW scatter (final output).
__global__ void gemm128_wmma(const __bf16* __restrict__ X,
                             const __bf16* __restrict__ Wp,
                             const float* __restrict__ bias,
                             float* __restrict__ outF,
                             __bf16* __restrict__ outB,
                             const float* __restrict__ gate,
                             int M, int mode) {
    __shared__ __bf16 sh[64 * Cc];           // 16 KB tile
    const int lane = threadIdx.x;
    const int j    = threadIdx.y;
    const int tid  = j * 32 + lane;
    const int m0   = blockIdx.x * 64;
    (void)M;

    // ---- async stage: 64 rows x 256 B = 1024 x 16 B chunks, coalesced ----
    {
        const char* gsrc = (const char*)(X + (size_t)m0 * Cc);
#pragma unroll
        for (int i = 0; i < 4; ++i) {
            int c = tid + 256 * i;
            unsigned loff = (unsigned)(unsigned long long)((const char*)sh + c * 16);
            async_g2l_b128(loff, gsrc + (size_t)c * 16);
        }
    }

    // ---- B fragments: load once, reuse across 4 M-subtiles ----
    v16bf bfr[4];
#pragma unroll
    for (int t = 0; t < 4; ++t)
        bfr[t] = *(const v16bf*)(Wp + ((size_t)(j * 4 + t) * 32 + lane) * 16);

    const int n    = j * 16 + (lane & 15);
    const float bv = bias[n];
    const int koff = (lane >> 4) * 8;
    const int lgrp = lane >> 4;
    float gs = 0.0f;
    if (mode == 2) gs = sigm(gate[n]);

    wait_asynccnt0();
    __syncthreads();

#pragma unroll
    for (int s = 0; s < 4; ++s) {
        const __bf16* xr = sh + (size_t)(s * 16 + (lane & 15)) * Cc;
        // hoist all 4 A fragments (8 ds_load_b128, clause-able, one wait)
        v16bf afr[4];
#pragma unroll
        for (int t = 0; t < 4; ++t) {
            v8bf lo = *(const v8bf*)(xr + t * 32 + koff);
            v8bf hi = *(const v8bf*)(xr + t * 32 + koff + 16);
            afr[t] = __builtin_shufflevector(lo, hi,
                        0,1,2,3,4,5,6,7,8,9,10,11,12,13,14,15);
        }
        v8f acc = {};
#pragma unroll
        for (int t = 0; t < 4; ++t)
            acc = __builtin_amdgcn_wmma_f32_16x16x32_bf16(false, afr[t], false, bfr[t],
                                                          (short)0, acc, false, false);
        const int rbase = m0 + s * 16 + (lgrp << 3);
        if (mode == 0) {
#pragma unroll
            for (int r = 0; r < 8; ++r)
                outF[(size_t)(rbase + r) * Cc + n] = acc[r] + bv;
        } else if (mode == 1) {
#pragma unroll
            for (int r = 0; r < 8; ++r) {
                float v = gelu_exact(acc[r] + bv);
                outB[(size_t)(rbase + r) * Cc + n] = (__bf16)v;
            }
        } else {
#pragma unroll
            for (int r = 0; r < 8; ++r) {
                int row = rbase + r;
                int b = row / HW, p = row % HW;
                outF[((size_t)b * Cc + n) * HW + p] = (acc[r] + bv) * gs;
            }
        }
    }
}

// --------------------------- depthwise 7x7 + GELU --------------------------
__global__ void dwconv_gelu(const float* __restrict__ q, const float* __restrict__ dww,
                            const float* __restrict__ dwb, __bf16* __restrict__ Xn) {
    int tid = blockIdx.x * blockDim.x + threadIdx.x;
    if (tid >= Bq * Cc * HW) return;
    int x = tid % Ww; int t = tid / Ww; int y = t % Hh; t /= Hh;
    int c = t % Cc; int b = t / Cc;
    const float* w   = dww + c * 49;
    const float* img = q + ((size_t)(b * Cc + c)) * HW;
    float s = 0.0f;
#pragma unroll
    for (int ky = 0; ky < 7; ++ky) {
        int yy = y + ky - 3;
        if (yy < 0 || yy >= Hh) continue;
#pragma unroll
        for (int kx = 0; kx < 7; ++kx) {
            int xx = x + kx - 3;
            if (xx < 0 || xx >= Ww) continue;
            s += img[yy * Ww + xx] * w[ky * 7 + kx];
        }
    }
    s = gelu_exact(s + dwb[c]);
    Xn[((size_t)b * HW + y * Ww + x) * Cc + c] = (__bf16)s;
}

// --------------------------- misc small kernels ----------------------------
__global__ void nearest_kernel(const float* __restrict__ rel_time, int* __restrict__ nidx) {
    int b = threadIdx.x;
    if (b >= Bq) return;
    float best = fabsf(rel_time[b * Nv]); int bi = 0;
    for (int n = 1; n < Nv; ++n) {
        float v = fabsf(rel_time[b * Nv + n]);
        if (v < best) { best = v; bi = n; }
    }
    nidx[b] = bi;
}

// correlation argmax over 5 shifts; initialize curr_offsets (repeat over P), attnP=0
__global__ void corr_init(const float* __restrict__ Yq, const float* __restrict__ Yv,
                          const int* __restrict__ nidx,
                          float* __restrict__ coff, float* __restrict__ attnP) {
    int m = blockIdx.x * blockDim.x + threadIdx.x;
    if (m >= ROWS) return;
    int b, hh, y, x; decode_m(m, b, hh, y, x);
    const float* qv = Yq + (((size_t)b * HW + y * Ww + x) * Cc + hh * HD);
    size_t vbase = (size_t)(b * Nv + nidx[b]) * HW;
    const int sdx[5] = {0, -1, 1, 0, 0};
    const int sdy[5] = {0, 0, 0, -1, 1};
    float best = -3.4e38f; int bi = 0;
#pragma unroll
    for (int s = 0; s < 5; ++s) {
        int ys = y - sdy[s], xs = x - sdx[s];
        float sc = 0.0f;
        if (ys >= 0 && ys < Hh && xs >= 0 && xs < Ww) {
            const float* vv = Yv + ((vbase + ys * Ww + xs) * Cc + hh * HD);
#pragma unroll
            for (int d = 0; d < HD; ++d) sc += qv[d] * vv[d];
        }
        if (sc > best) { best = sc; bi = s; }
    }
    // reference: offset = [bo_y*(2/W), bo_x*(2/H)] with bo=(dx,dy)
    float offx = sdy[bi] * (2.0f / Ww);
    float offy = sdx[bi] * (2.0f / Hh);
#pragma unroll
    for (int p = 0; p < Pp; ++p) {
        coff[((size_t)m * Pp + p) * 2 + 0] = offx;
        coff[((size_t)m * Pp + p) * 2 + 1] = offy;
        attnP[(size_t)m * Pp + p] = 0.0f;
    }
}

// bilinear-sample v_nearest at p0, pack GRU A-row [x(16)|p0(2)|h(16)|pad->64] bf16
__global__ void sample_pack(const float* __restrict__ Yv, const float* __restrict__ net,
                            const float* __restrict__ coff, const int* __restrict__ nidx,
                            __bf16* __restrict__ Ag) {
    int m = blockIdx.x * blockDim.x + threadIdx.x;
    if (m >= ROWS) return;
    int b, hh, y, x; decode_m(m, b, hh, y, x);
    float p0x = coff[(size_t)m * 8 + 0];
    float p0y = coff[(size_t)m * 8 + 1];
    float gx = -1.0f + 2.0f * x / (Ww - 1);
    float gy = -1.0f + 2.0f * y / (Hh - 1);
    float sx = fminf(fmaxf(gx + p0x, -1.0f), 1.0f);
    float sy = fminf(fmaxf(gy + p0y, -1.0f), 1.0f);
    float ix = (sx + 1.0f) * 0.5f * (Ww - 1);
    float iy = (sy + 1.0f) * 0.5f * (Hh - 1);
    float x0f = floorf(ix), y0f = floorf(iy);
    float wx = ix - x0f, wy = iy - y0f;
    int x0 = iclampi((int)x0f, 0, Ww - 1), x1 = iclampi((int)x0f + 1, 0, Ww - 1);
    int y0 = iclampi((int)y0f, 0, Hh - 1), y1 = iclampi((int)y0f + 1, 0, Hh - 1);
    size_t vbase = (size_t)(b * Nv + nidx[b]) * HW;
    const float* p00 = Yv + ((vbase + y0 * Ww + x0) * Cc + hh * HD);
    const float* p01 = Yv + ((vbase + y0 * Ww + x1) * Cc + hh * HD);
    const float* p10 = Yv + ((vbase + y1 * Ww + x0) * Cc + hh * HD);
    const float* p11 = Yv + ((vbase + y1 * Ww + x1) * Cc + hh * HD);
    float w00 = (1 - wx) * (1 - wy), w01 = wx * (1 - wy);
    float w10 = (1 - wx) * wy,       w11 = wx * wy;
    __bf16* arow = Ag + (size_t)m * 64;
    const float* nrow = net + (((size_t)b * HW + y * Ww + x) * Cc + hh * HD);
#pragma unroll
    for (int d = 0; d < HD; ++d) {
        float s = w00 * p00[d] + w01 * p01[d] + w10 * p10[d] + w11 * p11[d];
        arow[d]      = (__bf16)s;
        arow[18 + d] = (__bf16)nrow[d];
    }
    arow[16] = (__bf16)p0x;
    arow[17] = (__bf16)p0y;
#pragma unroll
    for (int d = 34; d < 64; ++d) arow[d] = (__bf16)0.0f;
}

// one wave per 16-row tile: async-stage the 16x64 A-tile to LDS, 12x WMMA ->
// [gi(48)|gh(48)], lane-local GRU nonlinearity, LDS-staged d_off/d_attn
// matvecs, accumulate updates.
__global__ void gru_step(const __bf16* __restrict__ Ag, const __bf16* __restrict__ Wgp,
                         const float* __restrict__ bih, const float* __restrict__ bhh,
                         const float* __restrict__ offw, const float* __restrict__ offb,
                         const float* __restrict__ aw,  const float* __restrict__ ab,
                         float* __restrict__ net, float* __restrict__ coff,
                         float* __restrict__ attnP) {
    __shared__ float  sh[16][17];
    __shared__ __bf16 ash[16 * 64];          // 2 KB A-tile
    const int lane = threadIdx.x;
    const int m0 = blockIdx.x * 16;

    // async stage 16 rows x 128 B = 128 x 16 B chunks
    {
        const char* gsrc = (const char*)(Ag + (size_t)m0 * 64);
#pragma unroll
        for (int i = 0; i < 4; ++i) {
            int c = lane + 32 * i;
            unsigned loff = (unsigned)(unsigned long long)((const char*)ash + c * 16);
            async_g2l_b128(loff, gsrc + (size_t)c * 16);
        }
    }

    v16bf bmat[6][2];
#pragma unroll
    for (int j = 0; j < 6; ++j)
#pragma unroll
        for (int t = 0; t < 2; ++t)
            bmat[j][t] = *(const v16bf*)(Wgp + ((size_t)(j * 2 + t) * 32 + lane) * 16);

    wait_asynccnt0();
    __syncthreads();

    const int koff = (lane >> 4) * 8;
    const __bf16* xr = ash + (size_t)(lane & 15) * 64;
    v16bf a[2];
#pragma unroll
    for (int t = 0; t < 2; ++t) {
        v8bf lo = *(const v8bf*)(xr + t * 32 + koff);
        v8bf hi = *(const v8bf*)(xr + t * 32 + koff + 16);
        a[t] = __builtin_shufflevector(lo, hi, 0,1,2,3,4,5,6,7,8,9,10,11,12,13,14,15);
    }
    v8f acc[6];
#pragma unroll
    for (int j = 0; j < 6; ++j) {
        acc[j] = (v8f){};
#pragma unroll
        for (int t = 0; t < 2; ++t)
            acc[j] = __builtin_amdgcn_wmma_f32_16x16x32_bf16(false, a[t], false,
                         bmat[j][t], (short)0, acc[j], false, false);
    }
    const int d = lane & 15;
    const int lgrp = lane >> 4;
    const float bir = bih[d], biz = bih[16 + d], bin = bih[32 + d];
    const float bhr = bhh[d], bhz = bhh[16 + d], bhn = bhh[32 + d];
#pragma unroll
    for (int r = 0; r < 8; ++r) {
        int m = m0 + r + 8 * lgrp;
        int b, hh, y, x; decode_m(m, b, hh, y, x);
        size_t ni = (((size_t)b * HW + y * Ww + x) * Cc + hh * HD + d);
        float hprev = net[ni];
        float rg = sigm(acc[0][r] + bir + acc[3][r] + bhr);
        float zg = sigm(acc[1][r] + biz + acc[4][r] + bhz);
        float ng = tanhf(acc[2][r] + bin + rg * (acc[5][r] + bhn));
        float nh = (1.0f - zg) * ng + zg * hprev;
        net[ni] = nh;
        sh[r + 8 * lgrp][d] = nh;
    }
    __syncthreads();
    for (int job = lane; job < 16 * 12; job += 32) {
        int ml = job / 12, o = job % 12;
        int m = m0 + ml;
        float s;
        if (o < 8) {
            const float* wr = offw + o * HD;
            s = offb[o];
#pragma unroll
            for (int dd = 0; dd < HD; ++dd) s += sh[ml][dd] * wr[dd];
            coff[((size_t)m * Pp + (o >> 1)) * 2 + (o & 1)] += s;
        } else {
            int oa = o - 8;
            const float* wr = aw + oa * HD;
            s = ab[oa];
#pragma unroll
            for (int dd = 0; dd < HD; ++dd) s += sh[ml][dd] * wr[dd];
            attnP[(size_t)m * Pp + oa] += s;
        }
    }
}

// softmax over (N,P), entropy per head, N*P bilinear gathers, weighted sum
__global__ void final_attn(const float* __restrict__ Yv, const float* __restrict__ coff,
                           const float* __restrict__ attnP,
                           const float* __restrict__ time_enc, const float* __restrict__ tw,
                           const float* __restrict__ tb,
                           __bf16* __restrict__ Xo, float* __restrict__ enth) {
    int m = blockIdx.x * blockDim.x + threadIdx.x;
    if (m >= ROWS) return;
    int b, hh, y, x; decode_m(m, b, hh, y, x);
    float tbias[Nv];
#pragma unroll
    for (int n = 0; n < Nv; ++n) {
        float s = tb[hh];
        const float* te = time_enc + (size_t)(b * Nv + n) * 64;
        const float* tr = tw + hh * 64;
        for (int k = 0; k < 64; ++k) s += te[k] * tr[k];
        tbias[n] = s;
    }
    float logits[16];
#pragma unroll
    for (int n = 0; n < Nv; ++n)
#pragma unroll
        for (int p = 0; p < Pp; ++p)
            logits[n * Pp + p] = attnP[(size_t)m * Pp + p] + tbias[n];
    float mx = logits[0];
#pragma unroll
    for (int i = 1; i < 16; ++i) mx = fmaxf(mx, logits[i]);
    float ssum = 0.0f, wgt[16];
#pragma unroll
    for (int i = 0; i < 16; ++i) { wgt[i] = __expf(logits[i] - mx); ssum += wgt[i]; }
    float inv = 1.0f / ssum, ent = 0.0f;
#pragma unroll
    for (int i = 0; i < 16; ++i) { wgt[i] *= inv; ent -= wgt[i] * logf(wgt[i] + 1e-8f); }
    enth[m] = ent;

    float accd[HD];
#pragma unroll
    for (int dd = 0; dd < HD; ++dd) accd[dd] = 0.0f;
    float gx = -1.0f + 2.0f * x / (Ww - 1);
    float gy = -1.0f + 2.0f * y / (Hh - 1);
    for (int p = 0; p < Pp; ++p) {
        float ox = coff[((size_t)m * Pp + p) * 2 + 0];
        float oy = coff[((size_t)m * Pp + p) * 2 + 1];
        float sx = fminf(fmaxf(gx + ox, -1.0f), 1.0f);
        float sy = fminf(fmaxf(gy + oy, -1.0f), 1.0f);
        float ix = (sx + 1.0f) * 0.5f * (Ww - 1);
        float iy = (sy + 1.0f) * 0.5f * (Hh - 1);
        float x0f = floorf(ix), y0f = floorf(iy);
        float wx = ix - x0f, wy = iy - y0f;
        int x0 = iclampi((int)x0f, 0, Ww - 1), x1 = iclampi((int)x0f + 1, 0, Ww - 1);
        int y0 = iclampi((int)y0f, 0, Hh - 1), y1 = iclampi((int)y0f + 1, 0, Hh - 1);
        float w00 = (1 - wx) * (1 - wy), w01 = wx * (1 - wy);
        float w10 = (1 - wx) * wy,       w11 = wx * wy;
        for (int n = 0; n < Nv; ++n) {
            size_t vb = (size_t)(b * Nv + n) * HW;
            const float* p00 = Yv + ((vb + y0 * Ww + x0) * Cc + hh * HD);
            const float* p01 = Yv + ((vb + y0 * Ww + x1) * Cc + hh * HD);
            const float* p10 = Yv + ((vb + y1 * Ww + x0) * Cc + hh * HD);
            const float* p11 = Yv + ((vb + y1 * Ww + x1) * Cc + hh * HD);
            float wnp = wgt[n * Pp + p];
#pragma unroll
            for (int dd = 0; dd < HD; ++dd)
                accd[dd] += wnp * (w00 * p00[dd] + w01 * p01[dd] +
                                   w10 * p10[dd] + w11 * p11[dd]);
        }
    }
    __bf16* orow = Xo + (((size_t)b * HW + y * Ww + x) * Cc + hh * HD);
#pragma unroll
    for (int dd = 0; dd < HD; ++dd) orow[dd] = (__bf16)accd[dd];
}

__global__ void ent_reduce(const float* __restrict__ enth,
                           float* __restrict__ out_conf, float* __restrict__ out_ent) {
    int pid = blockIdx.x * blockDim.x + threadIdx.x;
    if (pid >= Bq * HW) return;
    int b = pid / HW, p = pid % HW;
    float s = 0.0f;
#pragma unroll
    for (int hh = 0; hh < HEADS; ++hh) s += enth[(size_t)(b * HEADS + hh) * HW + p];
    s *= (1.0f / HEADS);
    const float maxent = logf((float)(Nv * Pp));
    float conf = 1.0f - fminf(fmaxf(s / (maxent + 1e-8f), 0.0f), 1.0f);
    out_ent[pid] = s;
    out_conf[pid] = conf;
}

// --------------------------------- launch ----------------------------------

extern "C" void kernel_launch(void* const* d_in, const int* in_sizes, int n_in,
                              void* d_out, int out_size, void* d_ws, size_t ws_size,
                              hipStream_t stream) {
    (void)in_sizes; (void)n_in; (void)out_size; (void)ws_size;
    const float* query    = (const float*)d_in[0];
    const float* values   = (const float*)d_in[1];
    const float* rel_time = (const float*)d_in[2];
    const float* time_enc = (const float*)d_in[3];
    const float* qw  = (const float*)d_in[4];  const float* qb  = (const float*)d_in[5];
    const float* vw  = (const float*)d_in[6];  const float* vb  = (const float*)d_in[7];
    const float* dww = (const float*)d_in[8];  const float* dwb = (const float*)d_in[9];
    const float* pww = (const float*)d_in[10]; const float* pwb = (const float*)d_in[11];
    const float* wih = (const float*)d_in[12]; const float* whh = (const float*)d_in[13];
    const float* bih = (const float*)d_in[14]; const float* bhh = (const float*)d_in[15];
    const float* offw= (const float*)d_in[16]; const float* offb= (const float*)d_in[17];
    const float* aw  = (const float*)d_in[18]; const float* ab  = (const float*)d_in[19];
    const float* tw  = (const float*)d_in[20]; const float* tb  = (const float*)d_in[21];
    const float* o1w = (const float*)d_in[22]; const float* o1b = (const float*)d_in[23];
    const float* o2w = (const float*)d_in[24]; const float* o2b = (const float*)d_in[25];
    const float* gate= (const float*)d_in[26];
    float* outp = (float*)d_out;

    char* ws = (char*)d_ws;
    // workspace layout (phase-disjoint buffers are aliased)
    __bf16* Xv  = (__bf16*)(ws + 0);          // 18,874,368 B ; aliased by Ag
    __bf16* Ag  = (__bf16*)(ws + 0);
    __bf16* Xq  = (__bf16*)(ws + 18874368);   //  4,718,592 B ; aliased by Xo
    __bf16* Xo  = Xq;
    __bf16* Xn  = (__bf16*)(ws + 23592960);   //  4,718,592 B ; aliased by Xo2
    __bf16* Xo2 = Xn;
    float*  Yq  = (float*)(ws + 28311552);    //  9,437,184 B
    float*  Yv  = (float*)(ws + 37748736);    // 37,748,736 B
    float*  net = (float*)(ws + 75497472);    //  9,437,184 B
    __bf16* Wq  = (__bf16*)(ws + 84934656);
    __bf16* Wv  = (__bf16*)(ws + 84967424);
    __bf16* Wpw = (__bf16*)(ws + 85000192);
    __bf16* W1  = (__bf16*)(ws + 85032960);
    __bf16* W2  = (__bf16*)(ws + 85065728);
    __bf16* Wg  = (__bf16*)(ws + 85098496);
    int*    nidx= (int*)  (ws + 85110784);
    float*  coff= (float*)(ws + 85111040);    //  4,718,592 B
    float*  attnP=(float*)(ws + 89829632);    //  2,359,296 B
    float*  enth= (float*)(ws + 92188928);    //    589,824 B

    // ---- weight + activation packing (bf16, WMMA fragment order) ----
    pack_w128<<<64, 256, 0, stream>>>(qw,  Wq);
    pack_w128<<<64, 256, 0, stream>>>(vw,  Wv);
    pack_w128<<<64, 256, 0, stream>>>(pww, Wpw);
    pack_w128<<<64, 256, 0, stream>>>(o1w, W1);
    pack_w128<<<64, 256, 0, stream>>>(o2w, W2);
    pack_wg  <<<24, 256, 0, stream>>>(wih, whh, Wg);
    pack_act <<<9216,  256, 0, stream>>>(query,  Xq, Bq);
    pack_act <<<36864, 256, 0, stream>>>(values, Xv, Bq * Nv);

    // ---- q/v projections (WMMA, 64-row tiles, async LDS staging) ----
    gemm128_wmma<<<PIX / 64,  dim3(32, 8), 0, stream>>>(Xq, Wq, qb, Yq, nullptr, nullptr, PIX,  0);
    gemm128_wmma<<<PIXV / 64, dim3(32, 8), 0, stream>>>(Xv, Wv, vb, Yv, nullptr, nullptr, PIXV, 0);

    // ---- depthwise 7x7 + GELU, then pointwise (WMMA) -> initial GRU hidden ----
    dwconv_gelu<<<9216, 256, 0, stream>>>(query, dww, dwb, Xn);
    gemm128_wmma<<<PIX / 64, dim3(32, 8), 0, stream>>>(Xn, Wpw, pwb, net, nullptr, nullptr, PIX, 0);

    // ---- nearest frame + correlation init ----
    nearest_kernel<<<1, 32, 0, stream>>>(rel_time, nidx);
    corr_init<<<ROWS / 256, 256, 0, stream>>>(Yq, Yv, nidx, coff, attnP);

    // ---- 3 GRU refinement iterations (WMMA gates) ----
    for (int it = 0; it < 3; ++it) {
        sample_pack<<<ROWS / 256, 256, 0, stream>>>(Yv, net, coff, nidx, Ag);
        gru_step<<<ROWS / 16, 32, 0, stream>>>(Ag, Wg, bih, bhh, offw, offb, aw, ab,
                                               net, coff, attnP);
    }

    // ---- softmax + deformable gather + weighted sum ----
    final_attn<<<ROWS / 256, 256, 0, stream>>>(Yv, coff, attnP, time_enc, tw, tb, Xo, enth);

    // ---- output MLP: o1 (GELU) then o2 (*sigmoid(gate)) -> NCHW f32 ----
    gemm128_wmma<<<PIX / 64, dim3(32, 8), 0, stream>>>(Xo,  W1, o1b, nullptr, Xo2, nullptr, PIX, 1);
    gemm128_wmma<<<PIX / 64, dim3(32, 8), 0, stream>>>(Xo2, W2, o2b, outp, nullptr, gate,   PIX, 2);

    // ---- confidence / entropy ----
    ent_reduce<<<Bq * HW / 256, 256, 0, stream>>>(enth, outp + (size_t)Bq * Cc * HW,
                                                  outp + (size_t)Bq * Cc * HW + Bq * HW);
}
